// Mamba2Block_88965952569954
// MI455X (gfx1250) — matrix operand order repaired and checked by reading, not compiled, not run on hardware
//
#include <hip/hip_runtime.h>
#include <hip/hip_bf16.h>
#include <math.h>

// ---------------------------------------------------------------------------
// Mamba2 block for MI455X (gfx1250, wave32, WMMA, async-to-LDS).
// Pipeline:
//   1. cvt weights f32->bf16
//   2. rmsnorm(x) -> u_bf16
//   3. WMMA GEMM1: zxbcdt = u @ W1^T     (bf16, f32 acc, 32x32 wave tile)
//   4. depthwise causal conv(4) + silu on xBC
//   5. selective scan per (b,h): state in VGPRs, B/C tiles double-buffered
//      into LDS via global_load_async_to_lds_b128 (ASYNCcnt pipeline)
//   6. y * silu(z), rmsnorm -> ybf16
//   7. WMMA GEMM2: out = x + ybf @ W2^T  (residual fused)
// ---------------------------------------------------------------------------

typedef __attribute__((ext_vector_type(16))) __bf16 v16bf;
typedef __attribute__((ext_vector_type(8)))  __bf16 v8bf;
typedef __attribute__((ext_vector_type(8)))  float  v8f;

static constexpr int Bb    = 4;
static constexpr int Ls    = 2048;
static constexpr int Dm    = 1024;              // d_model
static constexpr int Din   = 2048;              // d_inner
static constexpr int Nh    = 32;                // nheads
static constexpr int Hd    = 64;                // headdim
static constexpr int Dst   = 128;               // d_state
static constexpr int Dproj = 2 * Din + 2 * Dst + Nh;  // 4384
static constexpr int Cch   = Din + 2 * Dst;           // 2304
static constexpr int Rows  = Bb * Ls;                 // 8192

static __device__ __forceinline__ __bf16 f2bf(float f) {
    union { float f; unsigned u; } v; v.f = f;
    unsigned r = (v.u + 0x7FFFu + ((v.u >> 16) & 1u)) >> 16;
    union { unsigned short s; __bf16 b; } o; o.s = (unsigned short)r;
    return o.b;
}

static __device__ __forceinline__ float silu(float x) {
    return x / (1.0f + __expf(-x));
}

// LDS byte offset of a __shared__ object: on gfx1250 the flat shared aperture
// keeps the LDS offset in addr[31:0] (ISA 10.2 aperture mapping).
template <typename T>
static __device__ __forceinline__ unsigned lds_off(T* p) {
    return (unsigned)(uintptr_t)p;
}

// async DMA: copy 16 bytes per lane from global directly into LDS (ASYNCcnt)
static __device__ __forceinline__ void async_copy_b128(unsigned lds_byte,
                                                       const void* gaddr) {
    asm volatile("global_load_async_to_lds_b128 %0, %1, off"
                 :: "v"(lds_byte), "v"(gaddr) : "memory");
}
static __device__ __forceinline__ void wait_async0() {
    asm volatile("s_wait_asynccnt 0x0" ::: "memory");
}

// --------------------------- weight conversion -----------------------------
__global__ void cvt_f32_bf16(const float* __restrict__ in,
                             __bf16* __restrict__ out, int n) {
    int i = blockIdx.x * blockDim.x + threadIdx.x;
    if (i < n) out[i] = f2bf(in[i]);
}

// --------------------------- rmsnorm(x) -> u_bf16 --------------------------
__global__ __launch_bounds__(256)
void rmsnorm_x(const float* __restrict__ x, const float* __restrict__ w,
               __bf16* __restrict__ u) {
    int r = blockIdx.x;            // row in [0, Rows)
    int t = threadIdx.x;
    const float* xr = x + (size_t)r * Dm;
    float v[4]; float ss = 0.f;
    #pragma unroll
    for (int i = 0; i < 4; ++i) { v[i] = xr[t + 256 * i]; ss += v[i] * v[i]; }
    __shared__ float red[256];
    red[t] = ss; __syncthreads();
    for (int s = 128; s > 0; s >>= 1) {
        if (t < s) red[t] += red[t + s];
        __syncthreads();
    }
    float sc = rsqrtf(red[0] * (1.0f / Dm) + 1e-4f);
    #pragma unroll
    for (int i = 0; i < 4; ++i) {
        int c = t + 256 * i;
        u[(size_t)r * Dm + c] = f2bf(v[i] * sc * w[c]);
    }
}

// --------------------------- WMMA bf16 GEMM --------------------------------
// C[M][N] = A[M][K] * Bw[N][K]^T (+ resid).  N,K compile-time so all sub-tile
// addressing folds into 24-bit VMEM immediates off 2 base pointers.
// Block tile 512(M) x 32(N): 8 waves stacked in M, each wave 32x32 =
// 2(M)x2(N) sub-tiles -> 4 v_wmma per K-step fed by 8 b128 loads
// (2 loads/WMMA, A and B fragments each reused twice).
template <int N, int K>
__global__ __launch_bounds__(256)
void gemm_bf16_wmma(const __bf16* __restrict__ A,
                    const __bf16* __restrict__ Bw,
                    const float* __restrict__ resid,
                    float* __restrict__ C) {
    int wave = threadIdx.x >> 5;            // 0..7  (M direction)
    int lane = threadIdx.x & 31;
    int m0 = blockIdx.x * 512 + wave * 64;  // wave covers 64 rows? no: 2x16=32
    m0 = blockIdx.x * 512 + wave * 64;      // two 16-row tiles at +0, +16*? see below
    int n0 = blockIdx.y * 32;               // two 16-col tiles at +0, +16
    int lr = lane & 15;
    int lh = lane >> 4;                     // 0: low K-half lanes, 1: high

    // wave m-tiles at m0 and m0+16 .. but wave covers 64 rows (2 tiles of 16
    // with stride 32 would overlap other waves) -> use tiles m0+0, m0+32
    // so that 8 waves * 64 rows = 512 block rows, tiles at +0 and +32.
    v8f acc00 = {}, acc01 = {}, acc10 = {}, acc11 = {};
    const __bf16* Abase = A  + (size_t)(m0 + lr) * K;      // tile1 at +32*K
    const __bf16* Brow  = Bw + (size_t)(n0 + lr) * K;      // tile1 at +16*K
    const int ka = lh * 8;    // A: lanes16-31 take K+8  (ISA 16-bit A layout)
    const int kb = lh * 8;    // B: lanes16-31 take K+8 (same layout as A)

    for (int k = 0; k < K; k += 32) {
        v8bf a0lo = *(const v8bf*)(Abase + k + ka);
        v8bf a0hi = *(const v8bf*)(Abase + k + ka + 16);
        v8bf a1lo = *(const v8bf*)(Abase + 32 * K + k + ka);
        v8bf a1hi = *(const v8bf*)(Abase + 32 * K + k + ka + 16);
        v8bf b0lo = *(const v8bf*)(Brow + k + kb);
        v8bf b0hi = *(const v8bf*)(Brow + k + kb + 16);
        v8bf b1lo = *(const v8bf*)(Brow + 16 * K + k + kb);
        v8bf b1hi = *(const v8bf*)(Brow + 16 * K + k + kb + 16);
        v16bf af0, af1, bf0, bf1;
        #pragma unroll
        for (int j = 0; j < 8; ++j) {
            af0[j] = a0lo[j]; af0[j + 8] = a0hi[j];
            af1[j] = a1lo[j]; af1[j + 8] = a1hi[j];
            bf0[j] = b0lo[j]; bf0[j + 8] = b0hi[j];
            bf1[j] = b1lo[j]; bf1[j + 8] = b1hi[j];
        }
        acc00 = __builtin_amdgcn_wmma_f32_16x16x32_bf16(
            false, af0, false, bf0, (short)0, acc00, false, false);
        acc01 = __builtin_amdgcn_wmma_f32_16x16x32_bf16(
            false, af0, false, bf1, (short)0, acc01, false, false);
        acc10 = __builtin_amdgcn_wmma_f32_16x16x32_bf16(
            false, af1, false, bf0, (short)0, acc10, false, false);
        acc11 = __builtin_amdgcn_wmma_f32_16x16x32_bf16(
            false, af1, false, bf1, (short)0, acc11, false, false);
    }

    // store: C/D layout -> lane (lr,lh) element i is row m+lh*8+i, col n+lr
    const v8f* accs[4] = {&acc00, &acc01, &acc10, &acc11};
    #pragma unroll
    for (int t = 0; t < 4; ++t) {
        int mt = (t >> 1) * 32;             // A sub-tile offset (0 or 32)
        int nt = (t & 1) * 16;              // B sub-tile offset (0 or 16)
        int col = n0 + nt + lr;
        #pragma unroll
        for (int i = 0; i < 8; ++i) {
            int r = m0 + mt + lh * 8 + i;
            size_t idx = (size_t)r * N + col;
            float v = (*accs[t])[i];
            if (resid) v += resid[idx];
            C[idx] = v;
        }
    }
}

// ------------------ depthwise causal conv1d(4) + bias + silu ---------------
__global__ void conv_silu(const float* __restrict__ zx,
                          const float* __restrict__ cw,
                          const float* __restrict__ cb,
                          float* __restrict__ xc) {
    size_t idx = (size_t)blockIdx.x * blockDim.x + threadIdx.x;
    if (idx >= (size_t)Rows * Cch) return;
    int c = (int)(idx % Cch);
    size_t rl = idx / Cch;             // = b*Ls + l
    int l = (int)(rl % Ls);
    size_t bbase = rl - l;             // b*Ls
    float acc = cb[c];
    #pragma unroll
    for (int k = 0; k < 4; ++k) {
        int li = l - 3 + k;
        if (li >= 0)
            acc += zx[(bbase + li) * (size_t)Dproj + Din + c] * cw[c * 4 + k];
    }
    xc[idx] = silu(acc);
}

// --------------------------- selective scan --------------------------------
// One block per (b, h); 64 threads (2 waves); thread p keeps h[p][0..127] in
// VGPRs.  B_t / C_t (512B each) are DMA'd global->LDS with
// global_load_async_to_lds_b128, double-buffered: issue tile t+1, compute
// step t, s_wait_asynccnt 0 + barrier.
__global__ __launch_bounds__(64)
void ssm_scan(const float* __restrict__ zx,  const float* __restrict__ xc,
              const float* __restrict__ dt_bias, const float* __restrict__ A_log,
              const float* __restrict__ Dv, float* __restrict__ y) {
    int bh = blockIdx.x;
    int b = bh >> 5, h = bh & 31;
    int p  = threadIdx.x;                      // headdim row 0..63
    int wv = threadIdx.x >> 5;                 // wave 0 copies B, wave 1 copies C
    int ln = threadIdx.x & 31;
    float Acoef = -__expf(A_log[h]);
    float dtb   = dt_bias[h];
    float Dh    = Dv[h];

    __shared__ float bcs[2][2 * Dst];          // [buf][ B:0..127 | C:128..255 ]
    unsigned my_lds[2];
    my_lds[0] = lds_off(&bcs[0][wv * Dst + ln * 4]);
    my_lds[1] = lds_off(&bcs[1][wv * Dst + ln * 4]);

    float hst[Dst];
    #pragma unroll
    for (int n = 0; n < Dst; ++n) hst[n] = 0.f;

    const size_t rowbase = (size_t)b * Ls;
    // prologue: DMA tiles for l = 0 into buffer 0
    {
        const float* src0 = xc + rowbase * (size_t)Cch + Din + wv * Dst + ln * 4;
        async_copy_b128(my_lds[0], src0);
    }
    wait_async0();
    __syncthreads();

    for (int l = 0; l < Ls; ++l) {
        size_t r = rowbase + l;
        const float* xr = xc + r * (size_t)Cch;
        // issue DMA for step l+1 into the other buffer (overlaps compute)
        if (l + 1 < Ls) {
            const float* srcn = xr + Cch + Din + wv * Dst + ln * 4;
            async_copy_b128(my_lds[(l + 1) & 1], srcn);
        }

        float xsv   = xr[h * Hd + p];
        float dtraw = zx[r * (size_t)Dproj + (Dproj - Nh) + h] + dtb;
        float dts   = dtraw > 20.f ? dtraw : log1pf(__expf(dtraw));
        float dA    = __expf(dts * Acoef);
        float dtx   = dts * xsv;

        const float2* b2 = (const float2*)&bcs[l & 1][0];
        const float2* c2 = (const float2*)&bcs[l & 1][Dst];
        float yv = 0.f;
        #pragma unroll
        for (int i = 0; i < Dst / 2; ++i) {
            float2 bb = b2[i];                          // ds_load_b64 broadcast
            float2 cc = c2[i];
            hst[2*i]   = fmaf(dA, hst[2*i],   dtx * bb.x);
            yv         = fmaf(hst[2*i],   cc.x, yv);
            hst[2*i+1] = fmaf(dA, hst[2*i+1], dtx * bb.y);
            yv         = fmaf(hst[2*i+1], cc.y, yv);
        }
        y[r * (size_t)Din + h * Hd + p] = fmaf(Dh, xsv, yv);

        wait_async0();          // l+1 tiles landed in LDS
        __syncthreads();        // all threads done reading buf[l&1]
    }
}

// ------------------ y * silu(z), rmsnorm -> bf16 ---------------------------
__global__ __launch_bounds__(256)
void gate_rmsnorm(const float* __restrict__ y, const float* __restrict__ zx,
                  const float* __restrict__ w, __bf16* __restrict__ yb) {
    int r = blockIdx.x;
    int t = threadIdx.x;
    const float* yr = y  + (size_t)r * Din;
    const float* zr = zx + (size_t)r * Dproj;   // z = first d_inner of zxbcdt
    float g[8]; float ss = 0.f;
    #pragma unroll
    for (int i = 0; i < 8; ++i) {
        int c = t + 256 * i;
        float z = zr[c];
        float gv = yr[c] * silu(z);
        g[i] = gv; ss += gv * gv;
    }
    __shared__ float red[256];
    red[t] = ss; __syncthreads();
    for (int s = 128; s > 0; s >>= 1) {
        if (t < s) red[t] += red[t + s];
        __syncthreads();
    }
    float sc = rsqrtf(red[0] * (1.0f / Din) + 1e-5f);
    #pragma unroll
    for (int i = 0; i < 8; ++i) {
        int c = t + 256 * i;
        yb[(size_t)r * Din + c] = f2bf(g[i] * sc * w[c]);
    }
}

// ---------------------------------------------------------------------------
extern "C" void kernel_launch(void* const* d_in, const int* in_sizes, int n_in,
                              void* d_out, int out_size, void* d_ws, size_t ws_size,
                              hipStream_t stream) {
    const float* x        = (const float*)d_in[0];   // (4,2048,1024)
    const float* norm_w   = (const float*)d_in[1];   // (1024)
    const float* in_projw = (const float*)d_in[2];   // (4384,1024)
    const float* conv_w   = (const float*)d_in[3];   // (2304,4)
    const float* conv_b   = (const float*)d_in[4];   // (2304)
    const float* dt_bias  = (const float*)d_in[5];   // (32)
    const float* A_log    = (const float*)d_in[6];   // (32)
    const float* Dvec     = (const float*)d_in[7];   // (32)
    const float* ssm_w    = (const float*)d_in[8];   // (2048)
    const float* out_projw= (const float*)d_in[9];   // (1024,2048)
    float* out = (float*)d_out;

    // workspace carve-up (256B aligned slices)
    char* ws = (char*)d_ws;
    size_t off = 0;
    auto take = [&](size_t bytes) -> char* {
        char* p = ws + off;
        off = (off + bytes + 255) & ~(size_t)255;
        return p;
    };
    __bf16* w1b = (__bf16*)take((size_t)Dproj * Dm * 2);     //  9.0 MB
    __bf16* w2b = (__bf16*)take((size_t)Dm * Din * 2);       //  4.2 MB
    __bf16* ub  = (__bf16*)take((size_t)Rows * Dm * 2);      // 16.8 MB
    float*  zx  = (float*) take((size_t)Rows * Dproj * 4);   // 143.7 MB
    float*  xc  = (float*) take((size_t)Rows * Cch * 4);     // 75.5 MB
    float*  ybuf= (float*) take((size_t)Rows * Din * 4);     // 67.1 MB
    __bf16* yb  = (__bf16*)take((size_t)Rows * Din * 2);     // 33.6 MB
    (void)ws_size; (void)n_in; (void)in_sizes; (void)out_size;

    // 1. weights -> bf16
    {
        int n1 = Dproj * Dm, n2 = Dm * Din;
        cvt_f32_bf16<<<(n1 + 255) / 256, 256, 0, stream>>>(in_projw, w1b, n1);
        cvt_f32_bf16<<<(n2 + 255) / 256, 256, 0, stream>>>(out_projw, w2b, n2);
    }
    // 2. rmsnorm(x) -> u_bf16
    rmsnorm_x<<<Rows, 256, 0, stream>>>(x, norm_w, ub);
    // 3. GEMM1: zxbcdt[8192,4384] = u[8192,1024] @ W1[4384,1024]^T
    gemm_bf16_wmma<Dproj, Dm><<<dim3(Rows / 512, Dproj / 32), 256, 0, stream>>>(
        ub, w1b, nullptr, zx);
    // 4. depthwise conv + silu -> xc[8192,2304]
    {
        size_t tot = (size_t)Rows * Cch;
        conv_silu<<<(unsigned)((tot + 255) / 256), 256, 0, stream>>>(
            zx, conv_w, conv_b, xc);
    }
    // 5. selective scan -> ybuf[8192,2048] (includes +D*xs)
    ssm_scan<<<Bb * Nh, 64, 0, stream>>>(zx, xc, dt_bias, A_log, Dvec, ybuf);
    // 6. gate + rmsnorm -> yb (bf16)
    gate_rmsnorm<<<Rows, 256, 0, stream>>>(ybuf, zx, ssm_w, yb);
    // 7. GEMM2: out[8192,1024] = x + yb[8192,2048] @ W2[1024,2048]^T
    gemm_bf16_wmma<Dm, Din><<<dim3(Rows / 512, Dm / 32), 256, 0, stream>>>(
        yb, w2b, x, out);
}
